// GCN_3212635537778
// MI455X (gfx1250) — matrix operand order repaired and checked by reading, not compiled, not run on hardware
//
#include <hip/hip_runtime.h>
#include <hip/hip_bf16.h>

// ---------------- problem constants (fixed by the reference) ----------------
#define NN      51200      // nodes
#define EE      819200     // edges
#define FIN     400        // input features
#define KP1     416        // FIN padded to multiple of 32
#define HD      256        // hidden dim
#define NG      128        // graphs
#define NPG     400        // nodes per graph

typedef __attribute__((ext_vector_type(16))) __bf16 v16bf;
typedef __attribute__((ext_vector_type(8)))  float  v8f;

// Native FP32 global atomic add (avoid clang's CAS-loop fallback).
__device__ __forceinline__ void atom_add_f32(float* p, float v) {
    asm volatile("global_atomic_add_f32 %0, %1, off" :: "v"(p), "v"(v) : "memory");
}

// ---------------------------------------------------------------------------
__global__ void __launch_bounds__(256) k_zero_u32(unsigned* p, int n) {
    int t = blockIdx.x * 256 + threadIdx.x;
    if (t < n) p[t] = 0u;
}

__global__ void __launch_bounds__(256) k_degree(const int* __restrict__ dst,
                                                unsigned* __restrict__ deg, int E) {
    int t = blockIdx.x * 256 + threadIdx.x;
    if (t < E) atomicAdd(&deg[dst[t]], 1u);
}

__global__ void __launch_bounds__(256) k_dis(const unsigned* __restrict__ deg,
                                             float* __restrict__ dis, int n) {
    int t = blockIdx.x * 256 + threadIdx.x;
    if (t < n) dis[t] = rsqrtf(1.0f + (float)deg[t]);
}

// ---------------------------------------------------------------------------
// Pack A: f32 [N,K] (optional relu) -> bf16 [N,KP] row-major, zero padded.
template<int K, int KPAD, bool RELU>
__global__ void __launch_bounds__(256)
k_pack_rows(const float* __restrict__ X, __bf16* __restrict__ Xp, long long total) {
    long long t = (long long)blockIdx.x * 256 + threadIdx.x;
    if (t >= total) return;
    int row = (int)(t / KPAD);
    int k   = (int)(t % KPAD);
    float v = (k < K) ? X[(size_t)row * K + k] : 0.0f;
    if (RELU) v = fmaxf(v, 0.0f);
    Xp[t] = (__bf16)v;
}

// Pack B: f32 W[K,HD] -> bf16 in WMMA B-fragment order:
//   Wp[(c*HD + col)*32 + q]  where q in 0..31 maps to k = c*32 + q
// (fragment element j of lane(idx,half) is k = c*32 + 16*half + j, col = tn*16+idx,
//  i.e. each lane's 16 values are 32 contiguous bytes)
template<int K, int CH>
__global__ void __launch_bounds__(256)
k_pack_W(const float* __restrict__ W, __bf16* __restrict__ Wp) {
    int t = blockIdx.x * 256 + threadIdx.x;          // CH*HD*32 elements
    if (t >= CH * HD * 32) return;
    int q   = t & 31;
    int col = (t >> 5) & (HD - 1);
    int c   = t >> 13;                                // / (HD*32)
    int k   = c * 32 + q;
    float v = (k < K) ? W[(size_t)k * HD + col] : 0.0f;
    Wp[t] = (__bf16)v;
}

// ---------------------------------------------------------------------------
// GEMM: Y[NN,HD] = Xp[NN,KP](bf16) @ W(bf16, packed)  — f32 accumulate.
// One wave computes a 16x64 strip (4 WMMAs, A fragment reused).
// Grid sized exactly => EXEC all-1s at every WMMA (ISA requirement).
template<int KPAD>
__global__ void __launch_bounds__(128)
k_gemm_wmma(const __bf16* __restrict__ Xp, const __bf16* __restrict__ Wp,
            float* __restrict__ Y) {
    constexpr int CH = KPAD / 32;
    const int lane = threadIdx.x & 31;
    const int wave = threadIdx.x >> 5;
    const int wt   = blockIdx.x * 4 + wave;   // wave-tile id
    const int tm   = wt >> 2;                 // 16-row block (0..NN/16-1)
    const int tg   = wt & 3;                  // 64-col group
    const int half = lane >> 4;
    const int idx  = lane & 15;

    const __bf16* ap = Xp + (size_t)(tm * 16 + idx) * KPAD + 16 * half;
    const __bf16* bp = Wp + ((size_t)(tg * 64 + idx) * 32) + 16 * half;

    v8f acc0 = {}, acc1 = {}, acc2 = {}, acc3 = {};
#pragma unroll
    for (int c = 0; c < CH; ++c) {
        union { v16bf v; uint4 q[2]; } A, B0, B1, B2, B3;
        A.q[0]  = *(const uint4*)(ap + c * 32);
        A.q[1]  = *(const uint4*)(ap + c * 32 + 8);
        const __bf16* bc = bp + (size_t)c * HD * 32;
        B0.q[0] = *(const uint4*)(bc +    0);  B0.q[1] = *(const uint4*)(bc +    8);
        B1.q[0] = *(const uint4*)(bc +  512);  B1.q[1] = *(const uint4*)(bc +  520);
        B2.q[0] = *(const uint4*)(bc + 1024);  B2.q[1] = *(const uint4*)(bc + 1032);
        B3.q[0] = *(const uint4*)(bc + 1536);  B3.q[1] = *(const uint4*)(bc + 1544);
        acc0 = __builtin_amdgcn_wmma_f32_16x16x32_bf16(false, A.v, false, B0.v, (short)0, acc0, false, false);
        acc1 = __builtin_amdgcn_wmma_f32_16x16x32_bf16(false, A.v, false, B1.v, (short)0, acc1, false, false);
        acc2 = __builtin_amdgcn_wmma_f32_16x16x32_bf16(false, A.v, false, B2.v, (short)0, acc2, false, false);
        acc3 = __builtin_amdgcn_wmma_f32_16x16x32_bf16(false, A.v, false, B3.v, (short)0, acc3, false, false);
    }
    // C/D layout: VGPR r holds (M = 8*half + r, N = idx)
    float* y0 = Y + (size_t)(tm * 16 + 8 * half) * HD + tg * 64 + idx;
#pragma unroll
    for (int r = 0; r < 8; ++r) {
        y0[(size_t)r * HD +  0] = acc0[r];
        y0[(size_t)r * HD + 16] = acc1[r];
        y0[(size_t)r * HD + 32] = acc2[r];
        y0[(size_t)r * HD + 48] = acc3[r];
    }
}

// ---------------------------------------------------------------------------
// agg[i][j] = dis[i]^2 * h[i][j] + bias[j]   (self-loop + bias seed)
__global__ void __launch_bounds__(256)
k_agg_init(const float* __restrict__ h, const float* __restrict__ dis,
           const float* __restrict__ bias, float* __restrict__ agg, long long total) {
    long long t = (long long)blockIdx.x * 256 + threadIdx.x;
    if (t >= total) return;
    int i = (int)(t >> 8);
    int j = (int)(t & 255);
    float dd = dis[i];
    agg[t] = dd * dd * h[t] + bias[j];
}

// agg[dst] += dis[src]*dis[dst] * h[src]  — 64 lanes/edge, float4 per lane.
__global__ void __launch_bounds__(256)
k_edge_agg256(const float* __restrict__ h, const float* __restrict__ dis,
              const int* __restrict__ src, const int* __restrict__ dst,
              float* __restrict__ agg, int E) {
    long long t = (long long)blockIdx.x * 256 + threadIdx.x;
    int e = (int)(t >> 6);
    int f = ((int)t & 63) << 2;
    if (e >= E) return;
    int s = src[e], d = dst[e];
    float norm = dis[s] * dis[d];
    const float4 hv = *(const float4*)(h + (size_t)s * HD + f);
    float* ap = agg + (size_t)d * HD + f;
    atom_add_f32(ap + 0, hv.x * norm);
    atom_add_f32(ap + 1, hv.y * norm);
    atom_add_f32(ap + 2, hv.z * norm);
    atom_add_f32(ap + 3, hv.w * norm);
}

// ---------------------------------------------------------------------------
// Layer 3: y[i][0..1] = relu(h[i]) @ W3[256,2]
__global__ void __launch_bounds__(256)
k_lin3(const float* __restrict__ h, const float* __restrict__ W3,
       float* __restrict__ y, int n) {
    int i = blockIdx.x * 256 + threadIdx.x;
    if (i >= n) return;
    const float* row = h + (size_t)i * HD;
    float a0 = 0.0f, a1 = 0.0f;
    for (int k = 0; k < HD; k += 4) {
        float4 v = *(const float4*)(row + k);
        v.x = fmaxf(v.x, 0.0f); v.y = fmaxf(v.y, 0.0f);
        v.z = fmaxf(v.z, 0.0f); v.w = fmaxf(v.w, 0.0f);
        a0 += v.x * W3[2*k+0] + v.y * W3[2*k+2] + v.z * W3[2*k+4] + v.w * W3[2*k+6];
        a1 += v.x * W3[2*k+1] + v.y * W3[2*k+3] + v.z * W3[2*k+5] + v.w * W3[2*k+7];
    }
    y[2 * i]     = a0;
    y[2 * i + 1] = a1;
}

__global__ void __launch_bounds__(256)
k_init3(const float* __restrict__ h3, const float* __restrict__ dis,
        const float* __restrict__ b3, float* __restrict__ agg3, int total) {
    int t = blockIdx.x * 256 + threadIdx.x;
    if (t >= total) return;
    int i = t >> 1, c = t & 1;
    float dd = dis[i];
    agg3[t] = dd * dd * h3[t] + b3[c];
}

__global__ void __launch_bounds__(256)
k_edge3(const float* __restrict__ h3, const float* __restrict__ dis,
        const int* __restrict__ src, const int* __restrict__ dst,
        float* __restrict__ agg3, int E) {
    int e = blockIdx.x * 256 + threadIdx.x;
    if (e >= E) return;
    int s = src[e], d = dst[e];
    float norm = dis[s] * dis[d];
    atom_add_f32(agg3 + 2 * (size_t)d,     norm * h3[2 * s]);
    atom_add_f32(agg3 + 2 * (size_t)d + 1, norm * h3[2 * s + 1]);
}

// Mean pool over 400 nodes per graph: out[g][c]
__global__ void __launch_bounds__(256)
k_pool(const float* __restrict__ agg3, float* __restrict__ out) {
    int t = threadIdx.x;            // 256 = NG*2
    int g = t >> 1, c = t & 1;
    float s = 0.0f;
    const float* p = agg3 + (size_t)g * NPG * 2 + c;
    for (int i = 0; i < NPG; ++i) s += p[2 * i];
    out[t] = s * (1.0f / (float)NPG);
}

// ---------------------------------------------------------------------------
extern "C" void kernel_launch(void* const* d_in, const int* in_sizes, int n_in,
                              void* d_out, int out_size, void* d_ws, size_t ws_size,
                              hipStream_t stream) {
    const float* x    = (const float*)d_in[0];
    const int*   eidx = (const int*)  d_in[1];
    const float* W1   = (const float*)d_in[2];
    const float* b1   = (const float*)d_in[3];
    const float* W2   = (const float*)d_in[4];
    const float* b2   = (const float*)d_in[5];
    const float* W3   = (const float*)d_in[6];
    const float* b3   = (const float*)d_in[7];
    float* out = (float*)d_out;

    const int* src = eidx;
    const int* dst = eidx + EE;

    // workspace carve-out (all offsets 16B-aligned)
    char* ws = (char*)d_ws;
    float*    dis  = (float*)ws;                               // NN f32
    unsigned* deg  = (unsigned*)(ws + (size_t)NN * 4);         // NN u32
    float*    bufA = (float*)(ws + (size_t)2 * NN * 4);        // NN*HD f32
    float*    bufB = bufA + (size_t)NN * HD;                   // NN*HD f32
    float*    h3   = bufB + (size_t)NN * HD;                   // NN*2 f32
    float*    agg3 = h3   + (size_t)NN * 2;                    // NN*2 f32
    __bf16*   Wp1  = (__bf16*)(agg3 + (size_t)NN * 2);         // KP1/32*HD*32 bf16
    __bf16*   Wp2  = Wp1 + (size_t)(KP1 / 32) * HD * 32;       // HD/32*HD*32 bf16
    __bf16*   Xp   = Wp2 + (size_t)(HD / 32) * HD * 32;        // NN*KP1 bf16 (reused for layer-2 A)

    const int gemmBlocks = (NN / 16) * 4 / 4;                  // wave-tiles/4 = 3200
    const long long tot256 = (long long)NN * HD;
    const int initBlocks = (int)(tot256 / 256);
    const int edgeBlocks = (int)(((long long)EE * 64) / 256);

    // degrees -> dis
    k_zero_u32<<<(NN + 255) / 256, 256, 0, stream>>>(deg, NN);
    k_degree  <<<(EE + 255) / 256, 256, 0, stream>>>(dst, deg, EE);
    k_dis     <<<(NN + 255) / 256, 256, 0, stream>>>(deg, dis, NN);

    // weight packs (tiny)
    k_pack_W<FIN, KP1 / 32><<<((KP1 / 32) * HD * 32 + 255) / 256, 256, 0, stream>>>(W1, Wp1);
    k_pack_W<HD,  HD  / 32><<<((HD  / 32) * HD * 32 + 255) / 256, 256, 0, stream>>>(W2, Wp2);

    // layer 1: pack X -> bf16(416), WMMA GEMM, aggregation
    {
        long long tot = (long long)NN * KP1;
        k_pack_rows<FIN, KP1, false><<<(int)((tot + 255) / 256), 256, 0, stream>>>(x, Xp, tot);
    }
    k_gemm_wmma<KP1><<<gemmBlocks, 128, 0, stream>>>(Xp, Wp1, bufA);
    k_agg_init <<<initBlocks, 256, 0, stream>>>(bufA, dis, b1, bufB, tot256);
    k_edge_agg256<<<edgeBlocks, 256, 0, stream>>>(bufA, dis, src, dst, bufB, EE);

    // layer 2: pack relu(agg1) -> bf16(256) (overlaid on Xp), WMMA GEMM, aggregation
    {
        long long tot = (long long)NN * HD;
        k_pack_rows<HD, HD, true><<<(int)((tot + 255) / 256), 256, 0, stream>>>(bufB, Xp, tot);
    }
    k_gemm_wmma<HD><<<gemmBlocks, 128, 0, stream>>>(Xp, Wp2, bufA);
    k_agg_init <<<initBlocks, 256, 0, stream>>>(bufA, dis, b2, bufB, tot256);
    k_edge_agg256<<<edgeBlocks, 256, 0, stream>>>(bufA, dis, src, dst, bufB, EE);

    // layer 3 (relu fused), aggregation, mean pool
    k_lin3 <<<(NN + 255) / 256, 256, 0, stream>>>(bufB, W3, h3, NN);
    k_init3<<<(2 * NN + 255) / 256, 256, 0, stream>>>(h3, dis, b3, agg3, 2 * NN);
    k_edge3<<<(EE + 255) / 256, 256, 0, stream>>>(h3, dis, src, dst, agg3, EE);
    k_pool <<<1, 256, 0, stream>>>(agg3, out);
}
